// RNN_en_9526237462983
// MI455X (gfx1250) — compile-verified, matrix operand order checked by reading, not run on hardware
//
#include <hip/hip_runtime.h>
#include <math.h>

// Problem constants (match reference)
#define L_SEQ 4096
#define EDIM  512
#define HDIM  1024
#define G3    3072   // 3*H
#define PADW  40     // padded halves per LDS tile row (80B -> conflict-free-ish)

typedef __attribute__((ext_vector_type(16))) __bf16 v16bf;
typedef __attribute__((ext_vector_type(8)))  __bf16 v8bf;
typedef __attribute__((ext_vector_type(8)))  float  v8f;

// Low 32 bits of a generic pointer into LDS == byte offset within the WG's
// LDS allocation (ISA 10.2 aperture mapping) — what async-to-LDS wants.
__device__ __forceinline__ uint32_t lds_off32(const void* p)
{
    return (uint32_t)(uintptr_t)p;
}

// One 16-byte per-lane async copy global -> LDS (ASYNCcnt-tracked).
__device__ __forceinline__ void async_b128(uint32_t ldsaddr, const void* gaddr)
{
    asm volatile("global_load_async_to_lds_b128 %0, %1, off"
                 :: "v"(ldsaddr), "v"(gaddr) : "memory");
}

__device__ __forceinline__ void wait_async0()
{
    asm volatile("s_wait_asynccnt 0" ::: "memory");
}

// ---------------------------------------------------------------------------
// Build one 16x32 bf16 WMMA operand from an LDS tile (row-major, PADW stride).
// ISA 7.12.2 16-bit layout (wave32): lanes 0-15 hold K = 0..7 and 16..23,
// lanes 16-31 hold K = 8..15 and 24..31. Row = tile row (lid&15 supplied).
// ---------------------------------------------------------------------------
__device__ __forceinline__ v16bf lds_operand(const __bf16* base, int row, int lid)
{
    const int koff = (lid >> 4) << 3;
    const __bf16* rp = base + row * PADW + koff;
    v8bf lo = *reinterpret_cast<const v8bf*>(rp);
    v8bf hi = *reinterpret_cast<const v8bf*>(rp + 16);
    v16bf out;
#pragma unroll
    for (int i = 0; i < 8; ++i) { out[i] = lo[i]; out[i + 8] = hi[i]; }
    return out;
}

// Same layout, straight from a row-major global matrix (used by the scan).
__device__ __forceinline__ v16bf load_tile16x32(const __bf16* __restrict__ base,
                                                int ld, int row, int k0, int lid)
{
    const int koff = k0 + ((lid >> 4) << 3);
    const v8bf* p = reinterpret_cast<const v8bf*>(base + (size_t)row * ld + koff);
    v8bf lo = p[0];
    v8bf hi = p[2];           // +16 halves
    v16bf out;
#pragma unroll
    for (int i = 0; i < 8; ++i) { out[i] = lo[i]; out[i + 8] = hi[i]; }
    return out;
}

// ---------------------------------------------------------------------------
// C[M,N] = A[M,K](bf16) * W[N,K]^T(bf16) + bias[N]   (f32 accumulate/output)
// Block: 256 threads = 8 waves (2x4); wave tile 32x32 (2x2 WMMA); block 64x128.
// A/B tiles double-buffered in LDS, filled by async global->LDS copies that
// overlap the WMMA work on the previous K-chunk.
// ---------------------------------------------------------------------------
__global__ __launch_bounds__(256) void wmma_gemm_bias(
    const __bf16* __restrict__ A, const __bf16* __restrict__ W,
    const float* __restrict__ bias, float* __restrict__ C,
    int M, int N, int K)
{
    __shared__ __align__(16) __bf16 As[2][64 * PADW];
    __shared__ __align__(16) __bf16 Bs[2][128 * PADW];

    const int tid  = threadIdx.x;
    const int lid  = tid & 31;
    const int wave = tid >> 5;
    const int wm   = wave >> 2;           // 0..1
    const int wn   = wave & 3;            // 0..3
    const int m0   = blockIdx.y * 64;
    const int n0   = blockIdx.x * 128;

    // Stage assignment: thread -> (row, 16B segment). 3 async b128 per thread.
    const int srow = tid >> 2;            // 0..63
    const int sseg = (tid & 3) * 8;       // half offset 0/8/16/24

    auto stage = [&](int k0, int buf) {
        async_b128(lds_off32(&As[buf][srow * PADW + sseg]),
                   A + (size_t)(m0 + srow) * K + k0 + sseg);
        async_b128(lds_off32(&Bs[buf][srow * PADW + sseg]),
                   W + (size_t)(n0 + srow) * K + k0 + sseg);
        async_b128(lds_off32(&Bs[buf][(srow + 64) * PADW + sseg]),
                   W + (size_t)(n0 + srow + 64) * K + k0 + sseg);
    };

    v8f acc00 = {}, acc01 = {}, acc10 = {}, acc11 = {};

    const int nstages = K >> 5;           // K / 32
    stage(0, 0);
    for (int s = 0; s < nstages; ++s) {
        wait_async0();                    // own ASYNCcnt drained ...
        __syncthreads();                  // ... then whole block's data visible
        const int buf = s & 1;
        if (s + 1 < nstages) stage((s + 1) << 5, buf ^ 1);

        const __bf16* Ab = As[buf];
        const __bf16* Bb = Bs[buf];
        const int r = lid & 15;
        v16bf a0 = lds_operand(Ab, wm * 32 + r,      lid);
        v16bf a1 = lds_operand(Ab, wm * 32 + 16 + r, lid);
        v16bf b0 = lds_operand(Bb, wn * 32 + r,      lid);
        v16bf b1 = lds_operand(Bb, wn * 32 + 16 + r, lid);
        acc00 = __builtin_amdgcn_wmma_f32_16x16x32_bf16(false, a0, false, b0, (short)0, acc00, false, false);
        acc01 = __builtin_amdgcn_wmma_f32_16x16x32_bf16(false, a0, false, b1, (short)0, acc01, false, false);
        acc10 = __builtin_amdgcn_wmma_f32_16x16x32_bf16(false, a1, false, b0, (short)0, acc10, false, false);
        acc11 = __builtin_amdgcn_wmma_f32_16x16x32_bf16(false, a1, false, b1, (short)0, acc11, false, false);
        __syncthreads();                  // all reads done before buf re-staged
    }

    // D layout: VGPR r holds (M = r + 8*(lane>=16), N = lane%16)
    const int mr = (lid >> 4) << 3;
    const int nc = lid & 15;
    v8f accs[4] = {acc00, acc01, acc10, acc11};
#pragma unroll
    for (int ti = 0; ti < 2; ++ti) {
#pragma unroll
        for (int tj = 0; tj < 2; ++tj) {
            v8f a = accs[ti * 2 + tj];
            const int mb = m0 + wm * 32 + ti * 16 + mr;
            const int nb = n0 + wn * 32 + tj * 16 + nc;
            const float bi = bias[nb];
#pragma unroll
            for (int r = 0; r < 8; ++r)
                C[(size_t)(mb + r) * N + nb] = a[r] + bi;
        }
    }
}

// ---------------------------------------------------------------------------
// Persistent GRU scan: 1 workgroup, 1024 threads = 32 waves.
// Per step: gh = W_hh(3072x1024,bf16) @ h via WMMA (h replicated into all 16
// B columns -> all D columns identical), then gate math on 1024 threads.
// h kept in LDS as both f32 (gate math) and bf16 (WMMA operand).
// ---------------------------------------------------------------------------
__global__ __launch_bounds__(1024) void gru_scan_wmma(
    const float*  __restrict__ gi,     // L x 3072 precomputed x@Wih^T + bih
    const __bf16* __restrict__ Whh,    // 3072 x 1024
    const float*  __restrict__ bhh,    // 3072
    __bf16*       __restrict__ ys_bf,  // L x 1024 outputs (bf16, feeds next GEMM)
    float*        __restrict__ h_final,// 1024
    int seqlen)
{
    __shared__ __align__(16) float  h_f[HDIM];
    __shared__ __align__(16) __bf16 h_b[HDIM];
    __shared__ __align__(16) float  gh[G3];

    const int tid  = threadIdx.x;
    const int lid  = tid & 31;
    const int wave = tid >> 5;        // 0..31 ; each wave owns 6 groups of 16 rows

    h_f[tid] = 0.f;                   // blockDim == 1024 == HDIM
    h_b[tid] = (__bf16)0.f;
    __syncthreads();

    const v8bf* hb8 = reinterpret_cast<const v8bf*>(h_b);

    for (int t = 0; t < seqlen; ++t) {
        v8f acc[6] = {};

        for (int kc = 0; kc < HDIM; kc += 32) {
            const int koff = kc + ((lid >> 4) << 3);
            // B operand: h chunk replicated across all 16 columns (ds_load_b128)
            v8bf lo = hb8[koff >> 3];
            v8bf hi = hb8[(koff >> 3) + 2];
            v16bf bv;
#pragma unroll
            for (int i = 0; i < 8; ++i) { bv[i] = lo[i]; bv[i + 8] = hi[i]; }
#pragma unroll
            for (int g = 0; g < 6; ++g) {
                const int row = (wave * 6 + g) * 16 + (lid & 15);
                v16bf av = load_tile16x32(Whh, HDIM, row, kc, lid);
                acc[g] = __builtin_amdgcn_wmma_f32_16x16x32_bf16(
                    false, av, false, bv, (short)0, acc[g], false, false);
            }
        }

        // All D columns identical; lanes 0 (M=r) and 16 (M=r+8) commit to LDS.
        if ((lid & 15) == 0) {
            const int mr = (lid >> 4) << 3;
#pragma unroll
            for (int g = 0; g < 6; ++g) {
                const int rb = (wave * 6 + g) * 16 + mr;
#pragma unroll
                for (int r = 0; r < 8; ++r)
                    gh[rb + r] = acc[g][r] + bhh[rb + r];
            }
        }
        __syncthreads();

        {   // gate math: one thread per hidden unit
            const int j = tid;
            const float* git = gi + (size_t)t * G3;
            const float rg = 1.f / (1.f + __expf(-(git[j] + gh[j])));
            const float zg = 1.f / (1.f + __expf(-(git[HDIM + j] + gh[HDIM + j])));
            const float ng = tanhf(git[2 * HDIM + j] + rg * gh[2 * HDIM + j]);
            const float hn = (1.f - zg) * ng + zg * h_f[j];
            h_f[j] = hn;
            const __bf16 hb = (__bf16)hn;
            h_b[j] = hb;
            ys_bf[(size_t)t * HDIM + j] = hb;
        }
        __syncthreads();
    }
    h_final[tid] = h_f[tid];
}

// ---------------------------------------------------------------------------
// Embedding gather + max-norm (norm clamp), output bf16 for the first GEMM.
// ---------------------------------------------------------------------------
__global__ __launch_bounds__(256) void embed_maxnorm(
    const int* __restrict__ idx, const float* __restrict__ E,
    __bf16* __restrict__ emb_bf)
{
    __shared__ float red[256];
    __shared__ float s_scale;
    const int t = blockIdx.x;
    const int tid = threadIdx.x;
    const int row = idx[t];
    const float* e = E + (size_t)row * EDIM;
    const float e0 = e[tid], e1 = e[tid + 256];
    red[tid] = e0 * e0 + e1 * e1;
    __syncthreads();
    for (int s = 128; s > 0; s >>= 1) {
        if (tid < s) red[tid] += red[tid + s];
        __syncthreads();
    }
    if (tid == 0) {
        const float nrm = sqrtf(red[0]);
        s_scale = fminf(1.f, 1.f / fmaxf(nrm, 1e-7f));
    }
    __syncthreads();
    const float sc = s_scale;
    emb_bf[(size_t)t * EDIM + tid]       = (__bf16)(e0 * sc);
    emb_bf[(size_t)t * EDIM + tid + 256] = (__bf16)(e1 * sc);
}

__global__ void f32_to_bf16(const float* __restrict__ in, __bf16* __restrict__ out, int n)
{
    int i = blockIdx.x * blockDim.x + threadIdx.x;
    const int stride = gridDim.x * blockDim.x;
    for (; i < n; i += stride) out[i] = (__bf16)in[i];
}

// d_out = [ ys2[-1] (=h2) | h1 | h2 ]   (3072 f32)
__global__ void assemble_out(const float* __restrict__ h1, const float* __restrict__ h2,
                             float* __restrict__ out)
{
    const int j = blockIdx.x * blockDim.x + threadIdx.x;
    if (j < HDIM) {
        out[j]            = h2[j];
        out[HDIM + j]     = h1[j];
        out[2 * HDIM + j] = h2[j];
    }
}

// ---------------------------------------------------------------------------
extern "C" void kernel_launch(void* const* d_in, const int* in_sizes, int n_in,
                              void* d_out, int out_size, void* d_ws, size_t ws_size,
                              hipStream_t stream)
{
    const int*   idx  = (const int*)  d_in[0];
    const float* E    = (const float*)d_in[1];
    const float* Wih0 = (const float*)d_in[2];
    const float* Whh0 = (const float*)d_in[3];
    const float* bih0 = (const float*)d_in[4];
    const float* bhh0 = (const float*)d_in[5];
    const float* Wih1 = (const float*)d_in[6];
    const float* Whh1 = (const float*)d_in[7];
    const float* bih1 = (const float*)d_in[8];
    const float* bhh1 = (const float*)d_in[9];
    float* out = (float*)d_out;

    char* ws = (char*)d_ws;
    size_t off = 0;
    auto alloc = [&](size_t bytes) -> void* {
        void* p = ws + off;
        off = (off + bytes + 255) & ~(size_t)255;
        return p;
    };
    __bf16* emb_bf  = (__bf16*)alloc((size_t)L_SEQ * EDIM * 2);
    __bf16* Wih0_bf = (__bf16*)alloc((size_t)G3 * EDIM * 2);
    __bf16* Whh0_bf = (__bf16*)alloc((size_t)G3 * HDIM * 2);
    __bf16* Wih1_bf = (__bf16*)alloc((size_t)G3 * HDIM * 2);
    __bf16* Whh1_bf = (__bf16*)alloc((size_t)G3 * HDIM * 2);
    float*  gi0     = (float*) alloc((size_t)L_SEQ * G3 * 4);
    float*  gi1     = (float*) alloc((size_t)L_SEQ * G3 * 4);
    __bf16* ys1_bf  = (__bf16*)alloc((size_t)L_SEQ * HDIM * 2);
    __bf16* ys2_bf  = (__bf16*)alloc((size_t)L_SEQ * HDIM * 2);
    float*  h1f     = (float*) alloc(HDIM * 4);
    float*  h2f     = (float*) alloc(HDIM * 4);

    // Weights -> bf16 (resident in L2 afterwards: ~22 MB total)
    f32_to_bf16<<<512, 256, 0, stream>>>(Wih0, Wih0_bf, G3 * EDIM);
    f32_to_bf16<<<512, 256, 0, stream>>>(Whh0, Whh0_bf, G3 * HDIM);
    f32_to_bf16<<<512, 256, 0, stream>>>(Wih1, Wih1_bf, G3 * HDIM);
    f32_to_bf16<<<512, 256, 0, stream>>>(Whh1, Whh1_bf, G3 * HDIM);

    embed_maxnorm<<<L_SEQ, 256, 0, stream>>>(idx, E, emb_bf);

    dim3 gemm_grid(G3 / 128, L_SEQ / 64);
    // Layer 0: parallel input projection, then serial scan
    wmma_gemm_bias<<<gemm_grid, 256, 0, stream>>>(emb_bf, Wih0_bf, bih0, gi0, L_SEQ, G3, EDIM);
    gru_scan_wmma<<<1, 1024, 0, stream>>>(gi0, Whh0_bf, bhh0, ys1_bf, h1f, L_SEQ);
    // Layer 1
    wmma_gemm_bias<<<gemm_grid, 256, 0, stream>>>(ys1_bf, Wih1_bf, bih1, gi1, L_SEQ, G3, HDIM);
    gru_scan_wmma<<<1, 1024, 0, stream>>>(gi1, Whh1_bf, bhh1, ys2_bf, h2f, L_SEQ);

    assemble_out<<<4, 256, 0, stream>>>(h1f, h2f, out);
}
